// MyLSTMCell_11845519802781
// MI455X (gfx1250) — matrix operand order repaired and checked by reading, not compiled, not run on hardware
//
#include <hip/hip_runtime.h>
#include <hip/hip_bf16.h>

#define B_DIM  32768
#define IN_DIM 512
#define H_DIM  512
#define K_DIM  1024   // IN + H concatenated

#define BM 128
#define BN 64
#define BK 64
#define LDT 72        // padded LDS row stride (ushorts): 144B = conflict-free & 16B aligned
#define NSTAGES (K_DIM / BK)

typedef __attribute__((ext_vector_type(16))) __bf16        v16bf;
typedef __attribute__((ext_vector_type(8)))  float         v8f;
typedef __attribute__((ext_vector_type(4)))  unsigned int  v4u;
typedef __attribute__((ext_vector_type(4)))  unsigned short v4s;
typedef __attribute__((ext_vector_type(4)))  float         v4f;

__device__ __forceinline__ unsigned short f2bf(float f) {
  unsigned u = __float_as_uint(f);
  unsigned r = u + 0x7FFFu + ((u >> 16) & 1u);   // round-to-nearest-even
  return (unsigned short)(r >> 16);
}

__device__ __forceinline__ float sigmoid_(float x) {
  return 1.0f / (1.0f + __expf(-x));
}

__device__ __forceinline__ float tanh_(float x) {
  float e = __expf(2.0f * x);
  return (e - 1.0f) / (e + 1.0f);
}

// CDNA5 async global->LDS copy, 16B per lane, tracked by ASYNCcnt (no VGPR staging).
__device__ __forceinline__ void async_cp16(unsigned lds_off, const void* gptr) {
  asm volatile("global_load_async_to_lds_b128 %0, %1, off"
               :: "v"(lds_off), "v"(gptr)
               : "memory");
}

// ---- pack x|h -> bf16 A[B][1024] -------------------------------------------
__global__ __launch_bounds__(256) void pack_inputs(const float* __restrict__ x,
                                                   const float* __restrict__ h,
                                                   unsigned short* __restrict__ A) {
  int t = blockIdx.x * 256 + threadIdx.x;
  size_t e = (size_t)t * 4;
  int row = (int)(e >> 10);
  int col = (int)(e & 1023);
  const float* src = (col < 512) ? (x + (size_t)row * 512 + col)
                                 : (h + (size_t)row * 512 + (col - 512));
  v4f f = *(const v4f*)src;
  v4s o;
  o.x = f2bf(f.x); o.y = f2bf(f.y); o.z = f2bf(f.z); o.w = f2bf(f.w);
  *(v4s*)(A + e) = o;
}

// ---- pack weights -> bf16 W[4][1024][512]  (k-major, rows 0..511 = Wx) -----
__global__ __launch_bounds__(256) void pack_weights(
    const float* __restrict__ Wix, const float* __restrict__ Wih,
    const float* __restrict__ Wfx, const float* __restrict__ Wfh,
    const float* __restrict__ Wcx, const float* __restrict__ Wch,
    const float* __restrict__ Wox, const float* __restrict__ Woh,
    unsigned short* __restrict__ W) {
  int t = blockIdx.x * 256 + threadIdx.x;
  int e = t * 4;                    // index into [4][1024][512]
  int n = e & 511;
  int k = (e >> 9) & 1023;
  int g = e >> 19;
  bool hi = (k >= 512);
  const float* S;
  if (g == 0)      S = hi ? Wih : Wix;
  else if (g == 1) S = hi ? Wfh : Wfx;
  else if (g == 2) S = hi ? Wch : Wcx;
  else             S = hi ? Woh : Wox;
  int kk = k & 511;
  v4f f = *(const v4f*)(S + (size_t)kk * 512 + n);
  v4s o;
  o.x = f2bf(f.x); o.y = f2bf(f.y); o.z = f2bf(f.z); o.w = f2bf(f.w);
  *(v4s*)(W + e) = o;
}

// ---- combined bias[4][512] = b_gx + b_gh -----------------------------------
__global__ __launch_bounds__(256) void pack_bias(
    const float* __restrict__ bix, const float* __restrict__ bih,
    const float* __restrict__ bfx, const float* __restrict__ bfh,
    const float* __restrict__ bcx, const float* __restrict__ bch,
    const float* __restrict__ box, const float* __restrict__ boh,
    float* __restrict__ bias) {
  int t = blockIdx.x * 256 + threadIdx.x;  // 2048
  int n = t & 511;
  int g = t >> 9;
  const float* X; const float* Hh;
  if (g == 0)      { X = bix; Hh = bih; }
  else if (g == 1) { X = bfx; Hh = bfh; }
  else if (g == 2) { X = bcx; Hh = bch; }
  else             { X = box; Hh = boh; }
  bias[t] = X[n] + Hh[n];
}

// ---- fused 4-gate WMMA GEMM + LSTM cell update -----------------------------
__global__ __launch_bounds__(256) void lstm_wmma(
    const unsigned short* __restrict__ A,    // [B][1024] bf16
    const unsigned short* __restrict__ W,    // [4][1024][512] bf16
    const float* __restrict__ bias,          // [4][512]
    const float* __restrict__ c,             // [B][512]
    float* __restrict__ out)                 // h_next [B][512] then c_next [B][512]
{
  __shared__ unsigned short As[2][BM * LDT];        // 2 x 18432 B
  __shared__ unsigned short Bs[2][4 * BK * LDT];    // 2 x 36864 B  (total 108 KB)

  const int tid  = threadIdx.x;
  const int lane = tid & 31;
  const int wave = tid >> 5;
  const int wm   = wave >> 1;        // 0..3 : 32-row strip
  const int wn   = wave & 1;         // 0..1 : 32-col strip
  const int nblk = blockIdx.x;       // 0..7
  const int mblk = blockIdx.y;       // 0..255

  const int l15 = lane & 15;
  const int lh  = lane >> 4;

  // LDS byte offsets of the two buffers (low 32 bits of generic shared address)
  unsigned asOff[2], bsOff[2];
  asOff[0] = (unsigned)(unsigned long long)(void*)&As[0][0];
  asOff[1] = (unsigned)(unsigned long long)(void*)&As[1][0];
  bsOff[0] = (unsigned)(unsigned long long)(void*)&Bs[0][0];
  bsOff[1] = (unsigned)(unsigned long long)(void*)&Bs[1][0];

  v8f acc[4][2][2] = {};             // [gate][mi][ni]

  // --- async stage issue: 4 A-chunks + 8 B-chunks of 16B per thread ---------
  auto issue_stage = [&](int s, int buf) {
    int k0 = s * BK;
#pragma unroll
    for (int i = 0; i < 4; ++i) {
      int chunk = tid + i * 256;
      int row = chunk >> 3;
      int cc  = chunk & 7;
      async_cp16(asOff[buf] + (unsigned)(row * LDT + cc * 8) * 2,
                 A + (size_t)(mblk * BM + row) * K_DIM + k0 + cc * 8);
    }
#pragma unroll
    for (int i = 0; i < 8; ++i) {
      int chunk = tid + i * 256;
      int g   = chunk >> 9;
      int row = (chunk >> 3) & 63;
      int cc  = chunk & 7;
      async_cp16(bsOff[buf] + (unsigned)((g * BK + row) * LDT + cc * 8) * 2,
                 W + ((size_t)g * K_DIM + (k0 + row)) * H_DIM + nblk * BN + cc * 8);
    }
  };

  issue_stage(0, 0);

  // Cap unrolling: 2 bodies keep buf parity static (immediate LDS offsets) while
  // staying I$-resident and under 256 VGPRs (no s_set_vgpr_msb in the hot loop).
#pragma unroll 2
  for (int s = 0; s < NSTAGES; ++s) {
    int buf = s & 1;
    if (s + 1 < NSTAGES) {
      issue_stage(s + 1, buf ^ 1);               // prefetch next stage (other buffer)
      asm volatile("s_wait_asynccnt 0xc" ::: "memory");  // stage s done (12 still in flight)
    } else {
      asm volatile("s_wait_asynccnt 0x0" ::: "memory");
    }
    __syncthreads();                              // stage-s data visible to all waves

    const unsigned short* as = &As[buf][0];
    const unsigned short* bs = &Bs[buf][0];
#pragma unroll
    for (int ks = 0; ks < BK; ks += 32) {
      union { v16bf v; v4u q[2]; } fa[2], fb[4][2];
      // A frags: lane = M row; K chunks at 8*lh and 16+8*lh (ISA 16-bit A layout)
#pragma unroll
      for (int mi = 0; mi < 2; ++mi) {
        int row = wm * 32 + mi * 16 + l15;
        int kf  = ks + (lh << 3);
        fa[mi].q[0] = *(const v4u*)(as + row * LDT + kf);
        fa[mi].q[1] = *(const v4u*)(as + row * LDT + kf + 16);
      }
      // B frags: lane = N col; K chunks at 8*lh and 16+8*lh (same order as A)
      int krow = ks + (lh << 3);
#pragma unroll
      for (int g = 0; g < 4; ++g) {
#pragma unroll
        for (int ni = 0; ni < 2; ++ni) {
          int ncol = wn * 32 + ni * 16 + l15;
          const unsigned short* p = bs + (g * BK + krow) * LDT + ncol;
          unsigned short* d = (unsigned short*)&fb[g][ni];
#pragma unroll
          for (int j = 0; j < 8; ++j) {
            d[j]     = p[j * LDT];
            d[8 + j] = p[(16 + j) * LDT];
          }
        }
      }
#pragma unroll
      for (int g = 0; g < 4; ++g)
#pragma unroll
        for (int mi = 0; mi < 2; ++mi)
#pragma unroll
          for (int ni = 0; ni < 2; ++ni)
            acc[g][mi][ni] = __builtin_amdgcn_wmma_f32_16x16x32_bf16(
                false, fa[mi].v, false, fb[g][ni].v,
                (short)0, acc[g][mi][ni], false, false);
    }
    __syncthreads();                              // buffer free for reuse at stage s+2
  }

  // fused epilogue: gate nonlinearities + cell update, in registers
  const int row0 = mblk * BM + wm * 32;
  const int col0 = nblk * BN + wn * 32;
#pragma unroll
  for (int ni = 0; ni < 2; ++ni) {
    int n_g = col0 + ni * 16 + l15;
    float bi = bias[0 * H_DIM + n_g];
    float bf = bias[1 * H_DIM + n_g];
    float bg = bias[2 * H_DIM + n_g];
    float bo = bias[3 * H_DIM + n_g];
#pragma unroll
    for (int mi = 0; mi < 2; ++mi) {
#pragma unroll
      for (int v = 0; v < 8; ++v) {
        int m_g = row0 + mi * 16 + lh * 8 + v;   // C/D layout: M = vgpr + 8*(lane>>4)
        int idx = m_g * H_DIM + n_g;
        float gi = sigmoid_(acc[0][mi][ni][v] + bi);
        float gf = sigmoid_(acc[1][mi][ni][v] + bf);
        float gg = tanh_   (acc[2][mi][ni][v] + bg);
        float go = sigmoid_(acc[3][mi][ni][v] + bo);
        float cn = gf * c[idx] + gi * gg;
        out[idx] = go * tanh_(cn);
        out[B_DIM * H_DIM + idx] = cn;
      }
    }
  }
}

extern "C" void kernel_launch(void* const* d_in, const int* in_sizes, int n_in,
                              void* d_out, int out_size, void* d_ws, size_t ws_size,
                              hipStream_t stream) {
  const float* x    = (const float*)d_in[0];
  const float* h    = (const float*)d_in[1];
  const float* c    = (const float*)d_in[2];
  const float* W_ix = (const float*)d_in[3];
  const float* b_ix = (const float*)d_in[4];
  const float* W_ih = (const float*)d_in[5];
  const float* b_ih = (const float*)d_in[6];
  const float* W_fx = (const float*)d_in[7];
  const float* b_fx = (const float*)d_in[8];
  const float* W_fh = (const float*)d_in[9];
  const float* b_fh = (const float*)d_in[10];
  const float* W_cx = (const float*)d_in[11];
  const float* b_cx = (const float*)d_in[12];
  const float* W_ch = (const float*)d_in[13];
  const float* b_ch = (const float*)d_in[14];
  const float* W_ox = (const float*)d_in[15];
  const float* b_ox = (const float*)d_in[16];
  const float* W_oh = (const float*)d_in[17];
  const float* b_oh = (const float*)d_in[18];

  unsigned short* A  = (unsigned short*)d_ws;                 // 64 MB
  unsigned short* W  = A + (size_t)B_DIM * K_DIM;             // 4 MB
  float*          bb = (float*)(W + (size_t)4 * K_DIM * H_DIM); // 8 KB

  pack_inputs <<<(B_DIM * K_DIM) / 1024, 256, 0, stream>>>(x, h, A);
  pack_weights<<<(4 * K_DIM * H_DIM) / 1024, 256, 0, stream>>>(
      W_ix, W_ih, W_fx, W_fh, W_cx, W_ch, W_ox, W_oh, W);
  pack_bias   <<<(4 * H_DIM) / 256, 256, 0, stream>>>(
      b_ix, b_ih, b_fx, b_fh, b_cx, b_ch, b_ox, b_oh, bb);

  lstm_wmma<<<dim3(H_DIM / BN, B_DIM / BM), 256, 0, stream>>>(
      A, W, bb, c, (float*)d_out);
}
